// BiAttention_42202348651074
// MI455X (gfx1250) — matrix operand order repaired and checked by reading, not compiled, not run on hardware
//
#include <hip/hip_runtime.h>

#define B_   8
#define LI_  2048
#define LM_  2048
#define D_   256
#define OD_  (4*D_)
#define NEGC 1e30f

typedef __attribute__((ext_vector_type(16))) _Float16 v16h;
typedef __attribute__((ext_vector_type(8)))  _Float16 v8h;
typedef __attribute__((ext_vector_type(8)))  float    v8f;
typedef unsigned int  u32x4 __attribute__((ext_vector_type(4)));
typedef int           i32x4 __attribute__((ext_vector_type(4)));
typedef int           i32x8 __attribute__((ext_vector_type(8)));

// ---------- WMMA fragment loaders (CDNA5 16x16x32 f16 layouts) ----------
// A (16x32, MxK): lane m=lane&15 holds row m; grp=lane>>4 selects K-halves.
// Per lane: halves 0..7 -> K = 8*grp + h ; halves 8..15 -> K = 8*grp + 16 + h.
__device__ __forceinline__ v16h loadA(const _Float16* p) {
  v8h lo = *(const v8h*)(p);
  v8h hi = *(const v8h*)(p + 16);
  v16h r;
  #pragma unroll
  for (int h = 0; h < 8; ++h) { r[h] = lo[h]; r[h + 8] = hi[h]; }
  return r;
}
// B (32x16, KxN) stored as B^T rows [n][k]: lane n=lane&15 holds col n;
// Per lane: halves 0..7 -> K = 8*grp + h ; halves 8..15 -> K = 8*grp + 16 + h.
__device__ __forceinline__ v16h loadB(const _Float16* p) {
  v8h lo = *(const v8h*)(p);
  v8h hi = *(const v8h*)(p + 16);
  v16h r;
  #pragma unroll
  for (int h = 0; h < 8; ++h) { r[h] = lo[h]; r[h + 8] = hi[h]; }
  return r;
}

__device__ __forceinline__ float redmax16(float v) {
  #pragma unroll
  for (int m = 1; m < 16; m <<= 1) v = fmaxf(v, __shfl_xor(v, m, 32));
  return v;
}
__device__ __forceinline__ float redsum16(float v) {
  #pragma unroll
  for (int m = 1; m < 16; m <<= 1) v += __shfl_xor(v, m, 32);
  return v;
}

// ---------- TDM: async DMA of one 32x256 f16 K-tile into LDS ----------
// LDS rows padded: 256 halves data + 8 halves pad (pad after 128 DW, pad 4 DW)
// Descriptor per CDNA5 ISA.
__device__ __forceinline__ void tdm_load_ktile(const _Float16* gsrc, unsigned lds_off) {
  unsigned long long ga = (unsigned long long)(uintptr_t)gsrc;
  u32x4 g0;
  g0[0] = 1u;                                  // count=1 (valid), user mode
  g0[1] = lds_off;                             // lds_addr (bytes)
  g0[2] = (unsigned)ga;                        // global_addr[31:0]
  g0[3] = (unsigned)((ga >> 32) & 0x01ffffffu) // global_addr[56:32]
          | 0x80000000u;                       // type=2 ("image") at [127:126]
  i32x8 g1;
  g1[0] = (int)0x07910000u; // data_size=2B(1<<16), pad_enable(1<<20),
                            // pad_interval=128DW(6<<22), pad_amount=4DW(3<<25)
  g1[1] = (int)(256u << 16);   // tensor_dim0 = 256  (bits 79:48, low half)
  g1[2] = (int)(32u  << 16);   // tensor_dim1 = 32   (bits 111:80, low half)
  g1[3] = (int)(256u << 16);   // tile_dim0   = 256  (bits 127:112)
  g1[4] = 32;                  // tile_dim1   = 32   (bits 143:128)
  g1[5] = 256;                 // tensor_dim0_stride = 256 (bits 207:160 low)
  g1[6] = 0; g1[7] = 0;
  i32x4 gz = {0, 0, 0, 0};
#if defined(__clang_major__) && (__clang_major__ >= 23)
  i32x8 gz8 = {0, 0, 0, 0, 0, 0, 0, 0};
  __builtin_amdgcn_tensor_load_to_lds(g0, g1, gz, gz, gz8, 0);
#else
  __builtin_amdgcn_tensor_load_to_lds(g0, g1, gz, gz, 0);
#endif
}

// ---------- prep: f16 conversions ----------
__global__ void prep_convert(const float* __restrict__ input,
                             const float* __restrict__ memory,
                             const float* __restrict__ dscale,
                             _Float16* __restrict__ Qh, _Float16* __restrict__ Kh) {
  size_t i = (size_t)blockIdx.x * 256 + threadIdx.x;
  if (i >= (size_t)B_ * LI_ * D_) return;
  int d = (int)(i & (D_ - 1));
  Qh[i] = (_Float16)(input[i] * dscale[d]);
  Kh[i] = (_Float16)memory[i];
}

// ---------- prep: per-row dots + mask bias (1 wave per row) ----------
__global__ void prep_dots(const float* __restrict__ input,
                          const float* __restrict__ memory,
                          const int* __restrict__ mask,
                          const float* __restrict__ w_in,
                          const float* __restrict__ w_mem,
                          float* __restrict__ indot, float* __restrict__ bias) {
  int row  = blockIdx.x * 8 + (threadIdx.x >> 5);
  int lane = threadIdx.x & 31;
  const int NI = B_ * LI_;
  bool isin = row < NI;
  const float* src = isin ? (input + (size_t)row * D_)
                          : (memory + (size_t)(row - NI) * D_);
  const float* w = isin ? w_in : w_mem;
  float acc = 0.f;
  #pragma unroll
  for (int c = lane; c < D_; c += 32) acc += src[c] * w[c];
  #pragma unroll
  for (int m = 1; m < 32; m <<= 1) acc += __shfl_xor(acc, m, 32);
  if (lane == 0) {
    if (isin) indot[row] = acc;
    else {
      int r = row - NI;
      bias[r] = acc - NEGC * (1.0f - (float)mask[r]);
    }
  }
}

// ---------- main fused attention: TDM stage -> S=QK^T (WMMA) -> softmax -> O=PV (WMMA) ----------
__global__ __launch_bounds__(256, 1)
void bi_attn_main(const _Float16* __restrict__ Qh, const _Float16* __restrict__ Kh,
                  const float* __restrict__ indot, const float* __restrict__ bias,
                  float* __restrict__ rowmax, float* __restrict__ out) {
  constexpr int LDK = 264;   // halves; 528B row stride (16B aligned, matches TDM pad)
  constexpr int LDV = 40;    // halves; 80B row stride  (16B aligned)
  __shared__ _Float16 Ks[32 * LDK];   // K tile  [j=32][d=256]  (B^T rows for S GEMM)
  __shared__ _Float16 Vt[D_ * LDV];   // V tile^T [d=256][j=32] (B^T rows for PV GEMM)
  __shared__ _Float16 Ps[8 * 16 * 32];// per-wave P staging

  const int tid  = threadIdx.x;
  const int lane = tid & 31;
  const int wid  = tid >> 5;
  const int n16  = lane & 15;
  const int grp  = lane >> 4;

  const int b  = blockIdx.x >> 4;           // 16 blocks per batch
  const int i0 = (blockIdx.x & 15) * 128 + wid * 16;

  const _Float16* qbase = Qh + ((size_t)(b * LI_ + i0) + n16) * D_; // A row m=n16
  const float*    bb    = bias + (size_t)b * LM_;
  const unsigned  ks_lds = (unsigned)(uintptr_t)&Ks[0];

  float id8[8];
  #pragma unroll
  for (int v = 0; v < 8; ++v)
    id8[v] = indot[(size_t)b * LI_ + i0 + v + 8 * grp];

  v8f zero8 = {};
  v8f O[16];
  #pragma unroll
  for (int t = 0; t < 16; ++t) O[t] = zero8;
  float mrow[8], lrow[8];
  #pragma unroll
  for (int v = 0; v < 8; ++v) { mrow[v] = -3.0e38f; lrow[v] = 0.f; }

  for (int j0 = 0; j0 < LM_; j0 += 32) {
    __syncthreads();   // previous chunk's compute done before restage
    const _Float16* gK = Kh + ((size_t)b * LM_ + j0) * D_;

    // async TDM DMA of K tile into padded LDS rows (issued once, by wave 0)
    if (wid == 0) tdm_load_ktile(gK, ks_lds);

    // concurrently: stage transposed V tile: Vt[d][j] = K[j][d]
    {
      int d = tid;
      #pragma unroll
      for (int j = 0; j < 32; ++j) Vt[d * LDV + j] = gK[(size_t)j * D_ + d];
    }
    // prefetch next chunk's K tile into L2 while we compute this one
    if (j0 + 32 < LM_)
      __builtin_prefetch((const char*)(gK + 32 * D_) + tid * 64, 0, 0);

    if (wid == 0) __builtin_amdgcn_s_wait_tensorcnt(0);
    __syncthreads();

    // ---- S = Q K^T over d (8 K-steps of 32) ----
    v8f S0 = zero8, S1 = zero8;
    #pragma unroll
    for (int k = 0; k < 8; ++k) {
      v16h a  = loadA(qbase + 32 * k + 8 * grp);
      v16h b0 = loadB(&Ks[n16 * LDK + 32 * k + 8 * grp]);
      v16h b1 = loadB(&Ks[(16 + n16) * LDK + 32 * k + 8 * grp]);
      S0 = __builtin_amdgcn_wmma_f32_16x16x32_f16(false, a, false, b0, (short)0, S0, false, false);
      S1 = __builtin_amdgcn_wmma_f32_16x16x32_f16(false, a, false, b1, (short)0, S1, false, false);
    }

    // ---- online softmax (rows live across 16-lane N groups) ----
    float bj0 = bb[j0 + n16];
    float bj1 = bb[j0 + 16 + n16];
    float p0[8], p1[8], alpha[8];
    #pragma unroll
    for (int v = 0; v < 8; ++v) {
      float s0 = S0[v] + id8[v] + bj0;
      float s1 = S1[v] + id8[v] + bj1;
      float rm = redmax16(fmaxf(s0, s1));
      float mn = fmaxf(mrow[v], rm);
      alpha[v] = __expf(mrow[v] - mn);
      mrow[v]  = mn;
      p0[v] = __expf(s0 - mn);
      p1[v] = __expf(s1 - mn);
      float rs = redsum16(p0[v] + p1[v]);
      lrow[v] = lrow[v] * alpha[v] + rs;
    }
    #pragma unroll
    for (int t = 0; t < 16; ++t) {
      #pragma unroll
      for (int v = 0; v < 8; ++v) O[t][v] *= alpha[v];
    }

    // ---- re-layout P (C-frag -> A-frag) through per-wave LDS ----
    _Float16* pw = &Ps[wid * 512];
    #pragma unroll
    for (int v = 0; v < 8; ++v) {
      pw[(v + 8 * grp) * 32 + n16]      = (_Float16)p0[v];
      pw[(v + 8 * grp) * 32 + 16 + n16] = (_Float16)p1[v];
    }
    v16h ap = loadA(pw + n16 * 32 + 8 * grp);   // ordered via DScnt (same wave)

    // ---- O += P V : 16 d-tiles, K = j-chunk of 32 ----
    #pragma unroll
    for (int t = 0; t < 16; ++t) {
      v16h bv = loadB(&Vt[(t * 16 + n16) * LDV + 8 * grp]);
      O[t] = __builtin_amdgcn_wmma_f32_16x16x32_f16(false, ap, false, bv, (short)0, O[t], false, false);
    }
  }

  // epilogue: normalize and write output_one into out[:, D:2D); save row maxes
  float inv[8];
  #pragma unroll
  for (int v = 0; v < 8; ++v) inv[v] = 1.0f / lrow[v];
  #pragma unroll
  for (int t = 0; t < 16; ++t) {
    #pragma unroll
    for (int v = 0; v < 8; ++v) {
      size_t row = (size_t)b * LI_ + i0 + v + 8 * grp;
      out[row * OD_ + D_ + t * 16 + n16] = O[t][v] * inv[v];
    }
  }
  if (n16 == 0) {
    #pragma unroll
    for (int v = 0; v < 8; ++v)
      rowmax[(size_t)b * LI_ + i0 + v + 8 * grp] = mrow[v];
  }
}

// ---------- weight_two softmax over i + output_two[b][d] ----------
__global__ __launch_bounds__(256)
void w2_out2(const float* __restrict__ rowmax, const float* __restrict__ input,
             float* __restrict__ out2) {
  __shared__ float w2[LI_];
  __shared__ float red[256];
  int b = blockIdx.x, tid = threadIdx.x;
  const float* rm = rowmax + (size_t)b * LI_;
  float mx = -3.0e38f;
  for (int i = tid; i < LI_; i += 256) mx = fmaxf(mx, rm[i]);
  red[tid] = mx; __syncthreads();
  for (int s = 128; s > 0; s >>= 1) { if (tid < s) red[tid] = fmaxf(red[tid], red[tid + s]); __syncthreads(); }
  mx = red[0]; __syncthreads();
  float sm = 0.f;
  for (int i = tid; i < LI_; i += 256) { float e = __expf(rm[i] - mx); w2[i] = e; sm += e; }
  red[tid] = sm; __syncthreads();
  for (int s = 128; s > 0; s >>= 1) { if (tid < s) red[tid] += red[tid + s]; __syncthreads(); }
  float inv = 1.0f / red[0];
  float acc = 0.f;
  const float* ib = input + (size_t)b * LI_ * D_;
  for (int i = 0; i < LI_; ++i) acc += w2[i] * ib[(size_t)i * D_ + tid];
  out2[b * D_ + tid] = acc * inv;
}

// ---------- final assembly of the 4 output sections ----------
__global__ void finalize(const float* __restrict__ input, const float* __restrict__ out2,
                         float* __restrict__ out) {
  size_t idx = (size_t)blockIdx.x * 256 + threadIdx.x;  // over B*LI*D
  if (idx >= (size_t)B_ * LI_ * D_) return;
  int d = (int)(idx & (D_ - 1));
  size_t row = idx >> 8;           // b*LI + i
  int b = (int)(row >> 11);        // LI = 2048
  float x = input[idx];
  float* orow = out + row * OD_;
  float o1 = orow[D_ + d];
  orow[d] = x;
  orow[2 * D_ + d] = x * o1;
  orow[3 * D_ + d] = out2[b * D_ + d] * o1;
}

extern "C" void kernel_launch(void* const* d_in, const int* in_sizes, int n_in,
                              void* d_out, int out_size, void* d_ws, size_t ws_size,
                              hipStream_t stream) {
  (void)in_sizes; (void)n_in; (void)out_size; (void)ws_size;
  const float* input  = (const float*)d_in[0];
  const float* memory = (const float*)d_in[1];
  const int*   mask   = (const int*)d_in[2];
  const float* w_in   = (const float*)d_in[3];
  const float* w_mem  = (const float*)d_in[4];
  const float* dscale = (const float*)d_in[5];
  float* out = (float*)d_out;

  char* ws = (char*)d_ws;
  const size_t half_bytes = (size_t)B_ * LI_ * D_ * sizeof(_Float16);
  _Float16* Qh    = (_Float16*)ws;  ws += half_bytes;
  _Float16* Kh    = (_Float16*)ws;  ws += half_bytes;
  float* indot    = (float*)ws;     ws += (size_t)B_ * LI_ * sizeof(float);
  float* bias     = (float*)ws;     ws += (size_t)B_ * LM_ * sizeof(float);
  float* rowmax   = (float*)ws;     ws += (size_t)B_ * LI_ * sizeof(float);
  float* out2     = (float*)ws;

  const int nconv = B_ * LI_ * D_;
  prep_convert<<<nconv / 256, 256, 0, stream>>>(input, memory, dscale, Qh, Kh);
  prep_dots<<<(B_ * (LI_ + LM_)) / 8, 256, 0, stream>>>(input, memory, mask,
                                                        w_in, w_mem, indot, bias);
  bi_attn_main<<<B_ * (LI_ / 128), 256, 0, stream>>>(Qh, Kh, indot, bias, rowmax, out);
  w2_out2<<<B_, 256, 0, stream>>>(rowmax, input, out2);
  finalize<<<nconv / 256, 256, 0, stream>>>(input, out2, out);
}
